// ConcodeDecoder_88579405513340
// MI455X (gfx1250) — compile-verified, matrix operand order checked by reading, not run on hardware
//
#include <hip/hip_runtime.h>
#include <hip/hip_bf16.h>
#include <math.h>

// Problem constants (match reference)
#define BD   64      // batch
#define TT   256     // timesteps
#define SS   256     // S1 == S2
#define DD   512     // decoder size
#define EE   512     // embedding size
#define GATES 2048   // 4*DD
#define SEGSZ 32768  // 64*512 elements per concat segment

typedef __attribute__((ext_vector_type(16))) _Float16 v16h;
typedef __attribute__((ext_vector_type(8)))  _Float16 v8h;
typedef __attribute__((ext_vector_type(8)))  float    v8f;

// ---------------------------------------------------------------------------
// Fragment loader for 16-bit A/B operands of v_wmma_f32_16x16x32_f16.
// Source is a row-major [rows x ld] fp16 matrix; a fragment covers 16 rows
// (row0..row0+15) x 32 k (k0..k0+31). Per CDNA5 ISA layout:
//   lanes 0-15 : row = row0+lane,    elements 0..7 = k0+0..7,  8..15 = k0+16..23
//   lanes 16-31: row = row0+lane-16, elements 0..7 = k0+8..15, 8..15 = k0+24..31
// Both 8-element groups are contiguous -> two 16-byte loads.
// ---------------------------------------------------------------------------
__device__ inline v16h load_frag16(const _Float16* __restrict__ base, int ld,
                                   int row0, int k0, int lane) {
  const _Float16* p = base + (size_t)(row0 + (lane & 15)) * ld + k0
                      + ((lane & 16) ? 8 : 0);
  v8h lo = *(const v8h*)(p);
  v8h hi = *(const v8h*)(p + 16);
  v16h r;
#pragma unroll
  for (int i = 0; i < 8; ++i) { r[i] = lo[i]; r[i + 8] = hi[i]; }
  return r;
}

// ---------------------------------------------------------------------------
// C[64, N] = A * BT^T (+bias)(+tanh)
//   A : fp16, NSEG contiguous segments of [64,512] (concat along K, K=512*NSEG)
//   BT: fp16 [N, K] row-major (weights pre-transposed -> contiguous k loads)
// Block = 128 threads (4 waves). Wave w -> 16-col slab; blockIdx.y -> 32-row
// half of M=64 (2 accumulators per wave for B-fragment reuse).
// NSEG is a template parameter: compile-time trip count -> unroll/pipeline.
// ---------------------------------------------------------------------------
template <int NSEG, bool TANH>
__global__ void __launch_bounds__(128)
gemm64_wmma(const _Float16* __restrict__ A, const _Float16* __restrict__ BT,
            int N, const float* __restrict__ bias,
            float* __restrict__ Cf, long ldCf,
            _Float16* __restrict__ Ch0, _Float16* __restrict__ Ch1, int ldCh) {
  const int lane = threadIdx.x & 31;
  const int wave = threadIdx.x >> 5;
  const int n0 = blockIdx.x * 64 + wave * 16;
  const int m0 = blockIdx.y * 32;
  constexpr int K = NSEG * 512;

  v8f acc0 = {};
  v8f acc1 = {};
#pragma unroll
  for (int s = 0; s < NSEG; ++s) {
    const _Float16* __restrict__ As = A + s * SEGSZ;
#pragma unroll 4
    for (int kk = 0; kk < 512; kk += 32) {
      const int k0 = s * 512 + kk;
      v16h a0 = load_frag16(As, 512, m0,      kk, lane);
      v16h a1 = load_frag16(As, 512, m0 + 16, kk, lane);
      v16h b  = load_frag16(BT, K,   n0,      k0, lane);
      acc0 = __builtin_amdgcn_wmma_f32_16x16x32_f16(false, a0, false, b,
                                                    (short)0, acc0, false, false);
      acc1 = __builtin_amdgcn_wmma_f32_16x16x32_f16(false, a1, false, b,
                                                    (short)0, acc1, false, false);
    }
  }

  // C/D layout: lane&15 -> column, vgpr v -> row (v, or v+8 for lanes 16-31)
  const int n  = n0 + (lane & 15);
  const int mb = (lane & 16) ? 8 : 0;
  const float bs = bias ? bias[n] : 0.0f;
#pragma unroll
  for (int v = 0; v < 8; ++v) {
    float x0 = acc0[v] + bs;
    float x1 = acc1[v] + bs;
    if (TANH) { x0 = tanhf(x0); x1 = tanhf(x1); }
    const int r0 = m0 + mb + v;
    const int r1 = m0 + 16 + mb + v;
    if (Cf) {
      Cf[(size_t)r0 * ldCf + n] = x0;
      Cf[(size_t)r1 * ldCf + n] = x1;
    }
    if (Ch0) {
      Ch0[(size_t)r0 * ldCh + n] = (_Float16)x0;
      Ch0[(size_t)r1 * ldCh + n] = (_Float16)x1;
    }
    if (Ch1) {
      Ch1[(size_t)r0 * ldCh + n] = (_Float16)x0;
      Ch1[(size_t)r1 * ldCh + n] = (_Float16)x1;
    }
  }
}

// ---------------------------------------------------------------------------
// Batched attention (per-batch context matrices; VALU path).
// One block per batch element: scores (dot over D), softmax over S, weighted
// context sum -> fp16 (feeds next WMMA GEMM). Optionally emits scores [B,T,S].
// ---------------------------------------------------------------------------
__global__ void __launch_bounds__(256)
attn_batched(const float* __restrict__ qa, const float* __restrict__ ctx,
             const unsigned char* __restrict__ mask,
             _Float16* __restrict__ c16, float* __restrict__ sc_out1,
             float* __restrict__ sc_out2, int t) {
  __shared__ float q[DD];
  __shared__ float sc[SS];
  __shared__ float red[16];
  const int b   = blockIdx.x;
  const int tid = threadIdx.x;

  q[tid]       = qa[b * DD + tid];
  q[tid + 256] = qa[b * DD + tid + 256];
  __syncthreads();

  // score for s = tid
  const float* __restrict__ row = ctx + ((size_t)b * SS + tid) * DD;
  float s = 0.f;
#pragma unroll 4
  for (int d = 0; d < DD; d += 4)
    s += q[d] * row[d] + q[d + 1] * row[d + 1] +
         q[d + 2] * row[d + 2] + q[d + 3] * row[d + 3];
  if (!mask[b * SS + tid]) s = -1e9f;

  // softmax across 256 (8 waves of 32)
  float m = s;
  for (int o = 16; o > 0; o >>= 1) m = fmaxf(m, __shfl_xor(m, o, 32));
  if ((tid & 31) == 0) red[tid >> 5] = m;
  __syncthreads();
  float mx = fmaxf(fmaxf(fmaxf(red[0], red[1]), fmaxf(red[2], red[3])),
                   fmaxf(fmaxf(red[4], red[5]), fmaxf(red[6], red[7])));
  float e = expf(s - mx);
  float sum = e;
  for (int o = 16; o > 0; o >>= 1) sum += __shfl_xor(sum, o, 32);
  if ((tid & 31) == 0) red[8 + (tid >> 5)] = sum;
  __syncthreads();
  float tot = red[8] + red[9] + red[10] + red[11] +
              red[12] + red[13] + red[14] + red[15];
  float a = e / tot;
  sc[tid] = a;
  if (sc_out1) {
    const size_t so = ((size_t)b * TT + t) * SS + tid;
    sc_out1[so] = a;
    if (sc_out2) sc_out2[so] = a;
  }
  __syncthreads();

  // context: thread handles dims tid, tid+256
  float a0 = 0.f, a1 = 0.f;
  const float* __restrict__ cbase = ctx + (size_t)b * SS * DD;
  for (int ss2 = 0; ss2 < SS; ++ss2) {
    const float w = sc[ss2];
    a0 += w * cbase[ss2 * DD + tid];
    a1 += w * cbase[ss2 * DD + tid + 256];
  }
  c16[b * DD + tid]       = (_Float16)a0;
  c16[b * DD + tid + 256] = (_Float16)a1;
}

// ---------------------------------------------------------------------------
// LSTM cell elementwise (torch gate order i,f,g,o); writes h (fp16) into every
// concat slot that consumes it, and records h (fp32) into the parent buffer.
// ---------------------------------------------------------------------------
__global__ void __launch_bounds__(256)
lstm_cell_update(const float* __restrict__ g, float* __restrict__ c,
                 _Float16* __restrict__ h_x, _Float16* __restrict__ h_cs,
                 _Float16* __restrict__ h_c3, float* __restrict__ buf, int t) {
  const int idx = blockIdx.x * 256 + threadIdx.x;  // 64*512
  const int b = idx >> 9;
  const int d = idx & 511;
  const float* __restrict__ gb = g + (size_t)b * GATES;
  const float gi = gb[d], gf = gb[512 + d], gg = gb[1024 + d], go = gb[1536 + d];
  const float si = 1.f / (1.f + expf(-gi));
  const float sf = 1.f / (1.f + expf(-gf));
  const float so = 1.f / (1.f + expf(-go));
  const float c2 = sf * c[idx] + si * tanhf(gg);
  const float h2 = so * tanhf(c2);
  c[idx] = c2;
  const _Float16 h2h = (_Float16)h2;
  h_x[idx]  = h2h;
  h_cs[idx] = h2h;
  h_c3[idx] = h2h;
  buf[((size_t)b * TT + t) * DD + d] = h2;
}

// ---------------------------------------------------------------------------
// Per-step input gather: embeddings + parent state -> first 4 xcat segments.
// ---------------------------------------------------------------------------
__global__ void __launch_bounds__(256)
gather_embed(const int* __restrict__ nt, const int* __restrict__ pr,
             const int* __restrict__ ppr, const int* __restrict__ pidx,
             const float* __restrict__ nt_emb, const float* __restrict__ rule_emb,
             const float* __restrict__ buf, _Float16* __restrict__ xcat, int t) {
  const int b = blockIdx.x;
  const int i_nt = nt[b * TT + t];
  const int i_r  = pr[b * TT + t];
  const int i_pr = ppr[b * TT + t];
  const int i_p  = pidx[b * TT + t];
  const float* __restrict__ rn = nt_emb   + (size_t)i_nt * EE;
  const float* __restrict__ rr = rule_emb + (size_t)i_r  * EE;
  const float* __restrict__ rp = rule_emb + (size_t)i_pr * EE;
  const float* __restrict__ rb = buf + ((size_t)b * TT + i_p) * DD;
  for (int k = threadIdx.x; k < EE; k += 256) {
    xcat[0 * SEGSZ + b * EE + k] = (_Float16)rn[k];
    xcat[1 * SEGSZ + b * EE + k] = (_Float16)rr[k];
    xcat[2 * SEGSZ + b * EE + k] = (_Float16)rp[k];
    xcat[3 * SEGSZ + b * EE + k] = (_Float16)rb[k];
  }
}

// ---------------------------------------------------------------------------
// One-time prep: fp16 weight conversion / transposition, state init.
// ---------------------------------------------------------------------------
__global__ void __launch_bounds__(256)
prep_wcat(const float* __restrict__ Wih, const float* __restrict__ Whh,
          _Float16* __restrict__ Wcat) {
  const size_t idx = (size_t)blockIdx.x * 256 + threadIdx.x;  // 2048*2560
  const int n = (int)(idx / 2560);
  const int k = (int)(idx % 2560);
  const float v = (k < 2048) ? Wih[(size_t)n * 2048 + k]
                             : Whh[(size_t)n * 512 + (k - 2048)];
  Wcat[idx] = (_Float16)v;
}

__global__ void __launch_bounds__(256)
prep_transpose(const float* __restrict__ W /*[K,N]*/,
               _Float16* __restrict__ WT /*[N,K]*/, int K, int N) {
  const size_t idx = (size_t)blockIdx.x * 256 + threadIdx.x;  // N*K
  const int n = (int)(idx / K);
  const int k = (int)(idx % K);
  WT[idx] = (_Float16)W[(size_t)k * N + n];
}

__global__ void __launch_bounds__(256)
init_state(const float* __restrict__ h0, const float* __restrict__ c0,
           float* __restrict__ c, _Float16* __restrict__ h_x,
           _Float16* __restrict__ h_cs, _Float16* __restrict__ h_c3,
           float* __restrict__ buf) {
  const size_t idx = (size_t)blockIdx.x * 256 + threadIdx.x;  // 64*256*512
  buf[idx] = 0.f;
  if (idx < (size_t)BD * DD) {
    c[idx] = c0[idx];
    const _Float16 h = (_Float16)h0[idx];
    h_x[idx]  = h;
    h_cs[idx] = h;
    h_c3[idx] = h;
  }
}

// ---------------------------------------------------------------------------
extern "C" void kernel_launch(void* const* d_in, const int* in_sizes, int n_in,
                              void* d_out, int out_size, void* d_ws,
                              size_t ws_size, hipStream_t stream) {
  (void)in_sizes; (void)n_in; (void)out_size; (void)ws_size;

  const int*   nt       = (const int*)d_in[0];
  const int*   prules   = (const int*)d_in[1];
  const int*   parules  = (const int*)d_in[2];
  const int*   pidx     = (const int*)d_in[3];
  const float* src_ctx  = (const float*)d_in[4];
  const float* rest_ctx = (const float*)d_in[5];
  const unsigned char* src_mask  = (const unsigned char*)d_in[6];
  const unsigned char* rest_mask = (const unsigned char*)d_in[7];
  const float* h0   = (const float*)d_in[8];
  const float* c0   = (const float*)d_in[9];
  const float* nt_emb   = (const float*)d_in[10];
  const float* rule_emb = (const float*)d_in[11];
  const float* Wih  = (const float*)d_in[12];
  const float* Whh  = (const float*)d_in[13];
  const float* b_lstm = (const float*)d_in[14];
  const float* Wa_src = (const float*)d_in[15];
  const float* Wo_src = (const float*)d_in[16];
  const float* Wa_var = (const float*)d_in[17];
  const float* Wo_var = (const float*)d_in[18];
  const float* Wl  = (const float*)d_in[19];
  const float* bl  = (const float*)d_in[20];

  float* out_ao  = (float*)d_out;                       // [B,T,D]
  float* out_sc1 = out_ao + (size_t)BD * TT * DD;       // [B,T,S]
  float* out_sc2 = out_sc1 + (size_t)BD * TT * SS;      // [B,T,S]

  // Workspace bump allocator (all tensors L2-resident on MI455X).
  char*  ws  = (char*)d_ws;
  size_t off = 0;
  auto alloc = [&](size_t bytes) -> void* {
    void* p = ws + off;
    off = (off + bytes + 255) & ~(size_t)255;
    return p;
  };

  _Float16* Wcat16 = (_Float16*)alloc((size_t)GATES * 2560 * 2);  // [2048,2560]
  _Float16* WaS16  = (_Float16*)alloc((size_t)DD * DD * 2);       // [512,512]
  _Float16* WoS16  = (_Float16*)alloc((size_t)DD * 1024 * 2);     // [512,1024]
  _Float16* WaV16  = (_Float16*)alloc((size_t)DD * DD * 2);
  _Float16* WoV16  = (_Float16*)alloc((size_t)DD * 1024 * 2);
  _Float16* Wl16   = (_Float16*)alloc((size_t)DD * 1536 * 2);     // [512,1536]
  float*    buf    = (float*)alloc((size_t)BD * TT * DD * 4);     // parent states
  float*    c_f    = (float*)alloc((size_t)BD * DD * 4);
  float*    g_f    = (float*)alloc((size_t)BD * GATES * 4);
  float*    qa_f   = (float*)alloc((size_t)BD * DD * 4);
  // Contiguous concat buffers (single-pointer A operands; 32768 elems/segment)
  _Float16* xcat16 = (_Float16*)alloc((size_t)5 * SEGSZ * 2);  // [nt|r|pr|par|h]
  _Float16* catS16 = (_Float16*)alloc((size_t)2 * SEGSZ * 2);  // [c_src|h]
  _Float16* catV16 = (_Float16*)alloc((size_t)2 * SEGSZ * 2);  // [c_var|so]
  _Float16* cat3_16= (_Float16*)alloc((size_t)3 * SEGSZ * 2);  // [h|so|vo]

  _Float16* h_x  = xcat16 + 4 * SEGSZ;
  _Float16* h_cs = catS16 + 1 * SEGSZ;
  _Float16* h_c3 = cat3_16;            // also A operand for qa_src GEMM
  _Float16* so_v = catV16 + 1 * SEGSZ;
  _Float16* so_3 = cat3_16 + 1 * SEGSZ;  // also A operand for qa_var GEMM
  _Float16* vo_3 = cat3_16 + 2 * SEGSZ;
  _Float16* cS   = catS16;
  _Float16* cV   = catV16;

  // ---- one-time prep --------------------------------------------------
  prep_wcat<<<(GATES * 2560) / 256, 256, 0, stream>>>(Wih, Whh, Wcat16);
  prep_transpose<<<(DD * DD) / 256, 256, 0, stream>>>(Wa_src, WaS16, DD, DD);
  prep_transpose<<<(DD * 1024) / 256, 256, 0, stream>>>(Wo_src, WoS16, 1024, DD);
  prep_transpose<<<(DD * DD) / 256, 256, 0, stream>>>(Wa_var, WaV16, DD, DD);
  prep_transpose<<<(DD * 1024) / 256, 256, 0, stream>>>(Wo_var, WoV16, 1024, DD);
  prep_transpose<<<(DD * 1536) / 256, 256, 0, stream>>>(Wl, Wl16, 1536, DD);
  init_state<<<(BD * TT * DD) / 256, 256, 0, stream>>>(h0, c0, c_f, h_x, h_cs,
                                                       h_c3, buf);

  // ---- sequential timestep loop (graph-captured) ----------------------
  for (int t = 0; t < TT; ++t) {
    gather_embed<<<BD, 256, 0, stream>>>(nt, prules, parules, pidx, nt_emb,
                                         rule_emb, buf, xcat16, t);
    // g = [x,h] @ [Wih|Whh]^T + b_lstm   (K=2560, N=2048)
    gemm64_wmma<5, false><<<dim3(GATES / 64, 2), 128, 0, stream>>>(
        xcat16, Wcat16, GATES, b_lstm, g_f, (long)GATES, nullptr, nullptr, 0);
    lstm_cell_update<<<(BD * DD) / 256, 256, 0, stream>>>(g_f, c_f, h_x, h_cs,
                                                          h_c3, buf, t);

    // src attention: qa = h @ Wa_src ; attn ; src_out = tanh([c,h] @ Wo_src)
    gemm64_wmma<1, false><<<dim3(DD / 64, 2), 128, 0, stream>>>(
        h_c3, WaS16, DD, nullptr, qa_f, (long)DD, nullptr, nullptr, 0);
    attn_batched<<<BD, 256, 0, stream>>>(qa_f, src_ctx, src_mask, cS,
                                         nullptr, nullptr, t);
    gemm64_wmma<2, true><<<dim3(DD / 64, 2), 128, 0, stream>>>(
        catS16, WoS16, DD, nullptr, nullptr, 0, so_v, so_3, DD);

    // var attention over rest_context (scores are outputs 2 & 3)
    gemm64_wmma<1, false><<<dim3(DD / 64, 2), 128, 0, stream>>>(
        so_3, WaV16, DD, nullptr, qa_f, (long)DD, nullptr, nullptr, 0);
    attn_batched<<<BD, 256, 0, stream>>>(qa_f, rest_ctx, rest_mask, cV,
                                         out_sc1, out_sc2, t);
    gemm64_wmma<2, true><<<dim3(DD / 64, 2), 128, 0, stream>>>(
        catV16, WoV16, DD, nullptr, nullptr, 0, vo_3, nullptr, DD);

    // ao = tanh([h, src_out, var_out] @ Wl + bl) -> output[:, t, :]
    gemm64_wmma<3, true><<<dim3(DD / 64, 2), 128, 0, stream>>>(
        cat3_16, Wl16, DD, bl, out_ao + (size_t)t * DD, (long)(TT * DD),
        nullptr, nullptr, 0);
  }
}
